// MultiHeadAttention_11879879544649
// MI455X (gfx1250) — matrix operand order repaired and checked by reading, not compiled, not run on hardware
//
#include <hip/hip_runtime.h>

// ---------------------------------------------------------------------------
// Types for CDNA5 WMMA (wave32): v_wmma_f32_16x16x32_bf16
// ---------------------------------------------------------------------------
typedef __attribute__((ext_vector_type(16))) __bf16 v16bf;
typedef __attribute__((ext_vector_type(8)))  __bf16 v8bf;
typedef __attribute__((ext_vector_type(8)))  float  v8f;

union BfFrag { v16bf v; v8bf h[2]; };

__device__ __forceinline__ v8f wmma_bf16(v16bf a, v16bf b, v8f c) {
  // 8 args: (neg_a, A, neg_b, B, c_mod, C, reuse_a, reuse_b)
  return __builtin_amdgcn_wmma_f32_16x16x32_bf16(false, a, false, b,
                                                 (short)0, c, false, false);
}

__device__ __forceinline__ v8f zero8() {
  v8f z;
#pragma unroll
  for (int i = 0; i < 8; ++i) z[i] = 0.0f;
  return z;
}

// Pack two f32 -> packed bf16 pair, 4-byte store (lowers to v_cvt_pk_bf16_f32).
__device__ __forceinline__ void store_pk_bf16(__bf16* dst, float a, float b) {
  union { __bf16 h[2]; unsigned u; } p;
  p.h[0] = (__bf16)a;
  p.h[1] = (__bf16)b;
  *(unsigned*)dst = p.u;
}

// A-matrix fragment (16x32 bf16, row-major, stride in elements).
// Layout: lanes 0-15 = rows, lanes 0-15 hold K 0..15, lanes 16-31 hold K 16..31.
__device__ __forceinline__ v16bf load_afrag(const __bf16* p, int stride) {
  int lane = threadIdx.x & 31;
  int m = lane & 15, hf = lane >> 4;
  const __bf16* src = p + (size_t)m * stride + 16 * hf;
  BfFrag f;
  f.h[0] = *(const v8bf*)(src);
  f.h[1] = *(const v8bf*)(src + 8);
  return f.v;
}

// B-matrix fragment (32x16 bf16) from [col][k] storage (stride in elements).
// Layout (same k order as load_afrag): lane = column, lanes 0-15 hold K 0..15, lanes 16-31 hold K 16..31.
__device__ __forceinline__ v16bf load_bfrag(const __bf16* p, int stride) {
  int lane = threadIdx.x & 31;
  int c = lane & 15, hf = lane >> 4;
  const __bf16* src = p + (size_t)c * stride + 16 * hf;
  BfFrag f;
  f.h[0] = *(const v8bf*)(src);
  f.h[1] = *(const v8bf*)(src + 8);
  return f.v;
}

// B fragment from emb with per-lane row clamp (relative distance may be <0
// near the causal diagonal; those entries are masked later).
__device__ __forceinline__ v16bf load_bfrag_eclamp(const __bf16* eb, int row0,
                                                   int koff) {
  int lane = threadIdx.x & 31;
  int c = lane & 15, hf = lane >> 4;
  int r = row0 + c;
  r = r < 0 ? 0 : (r > 2047 ? 2047 : r);
  const __bf16* src = eb + (size_t)r * 64 + koff + 16 * hf;
  BfFrag f;
  f.h[0] = *(const v8bf*)(src);
  f.h[1] = *(const v8bf*)(src + 8);
  return f.v;
}

// ---------------------------------------------------------------------------
// Kernel 1: positional embedding  emb[n][s][h] = sqrt(2)/16 * sum_f sin/cos * P
// ---------------------------------------------------------------------------
__global__ __launch_bounds__(64) void emb_kernel(const float* __restrict__ P,
                                                 __bf16* __restrict__ eb) {
  int pos  = blockIdx.x & 2047;
  int head = blockIdx.x >> 11;
  int hh   = threadIdx.x;
  float acc = 0.0f;
  for (int f2 = 0; f2 < 128; ++f2) {
    // inv_freq = 10000^(-(2*f2)/256)
    float invf = expf(-(2.0f * (float)f2 / 256.0f) * 9.210340372f);
    float ang = (float)pos * invf;
    float sv = sinf(ang), cv = cosf(ang);
    acc += sv * P[((size_t)(2 * f2) * 16 + head) * 64 + hh];
    acc += cv * P[((size_t)(2 * f2 + 1) * 16 + head) * 64 + hh];
  }
  // sqrt(2) * FREQS^-0.5 = 1.41421356/16
  eb[((size_t)head * 2048 + pos) * 64 + hh] = (__bf16)(acc * 0.0883883476f);
}

// ---------------------------------------------------------------------------
// Kernel 2: QKV projection  (4096 x 1024) x (1024 x 3072), bf16 WMMA
//   K-step 64: 8 WMMAs per barrier pair. A tile 128x64, B tile 64x64 (24 KB).
//   q -> [b][n][s][h]*h^-0.5 (+bias), k -> [b][n][s][h], v -> [b][n][h][s]
// ---------------------------------------------------------------------------
__global__ __launch_bounds__(256) void qkv_gemm(const float* __restrict__ X,
                                                const float* __restrict__ W,
                                                const float* __restrict__ qbias,
                                                __bf16* __restrict__ qb,
                                                __bf16* __restrict__ kb,
                                                __bf16* __restrict__ vb) {
  __shared__ __attribute__((aligned(16))) __bf16 As[128 * 64];
  __shared__ __attribute__((aligned(16))) __bf16 Bs[64 * 64];
  int m0 = blockIdx.x * 128, n0 = blockIdx.y * 64;
  int t = threadIdx.x;
  int w = t >> 5, wm = w >> 1, wn = w & 1;
  v8f acc[2][2];
#pragma unroll
  for (int i = 0; i < 2; ++i)
#pragma unroll
    for (int j = 0; j < 2; ++j) acc[i][j] = zero8();

  for (int k0 = 0; k0 < 1024; k0 += 64) {
    {  // stage A tile 128x64 (f32 -> bf16), packed 16-byte LDS stores
      int row = t >> 1, seg = (t & 1) * 32;
      const float* src = X + (size_t)(m0 + row) * 1024 + k0 + seg;
      __bf16* dst = As + row * 64 + seg;
#pragma unroll
      for (int v = 0; v < 4; ++v) {
        v8bf tmp;
#pragma unroll
        for (int i = 0; i < 8; ++i) tmp[i] = (__bf16)src[v * 8 + i];
        *(v8bf*)(dst + v * 8) = tmp;
      }
    }
    {  // stage B tile 64x64 transposed to [n][k], packed b32 LDS stores
      int nb = (t & 15) * 4, kb4 = (t >> 4) * 4;
#pragma unroll
      for (int i = 0; i < 4; ++i)
#pragma unroll
        for (int j = 0; j < 2; ++j) {
          float lo = W[(size_t)(k0 + kb4 + 2 * j) * 3072 + n0 + nb + i];
          float hi = W[(size_t)(k0 + kb4 + 2 * j + 1) * 3072 + n0 + nb + i];
          store_pk_bf16(Bs + (nb + i) * 64 + kb4 + 2 * j, lo, hi);
        }
    }
    __syncthreads();
    v16bf a00 = load_afrag(As + (wm * 32 + 0) * 64 + 0, 64);
    v16bf a01 = load_afrag(As + (wm * 32 + 0) * 64 + 32, 64);
    v16bf a10 = load_afrag(As + (wm * 32 + 16) * 64 + 0, 64);
    v16bf a11 = load_afrag(As + (wm * 32 + 16) * 64 + 32, 64);
    v16bf b00 = load_bfrag(Bs + (wn * 32 + 0) * 64 + 0, 64);
    v16bf b01 = load_bfrag(Bs + (wn * 32 + 0) * 64 + 32, 64);
    v16bf b10 = load_bfrag(Bs + (wn * 32 + 16) * 64 + 0, 64);
    v16bf b11 = load_bfrag(Bs + (wn * 32 + 16) * 64 + 32, 64);
    acc[0][0] = wmma_bf16(a00, b00, acc[0][0]);
    acc[0][0] = wmma_bf16(a01, b01, acc[0][0]);
    acc[0][1] = wmma_bf16(a00, b10, acc[0][1]);
    acc[0][1] = wmma_bf16(a01, b11, acc[0][1]);
    acc[1][0] = wmma_bf16(a10, b00, acc[1][0]);
    acc[1][0] = wmma_bf16(a11, b01, acc[1][0]);
    acc[1][1] = wmma_bf16(a10, b10, acc[1][1]);
    acc[1][1] = wmma_bf16(a11, b11, acc[1][1]);
    __syncthreads();
  }

  // Epilogue. n-tiles are 64-wide and aligned, so the q/k/v selector and the
  // head index are uniform for the whole block -> single uniform branch.
  int lane = t & 31, c = lane & 15, g = lane >> 4;
  const float wscale = 0.0237448648f;  // (3*1024*64*16)^-0.25
  int atype = n0 >> 10;
  int head = (n0 >> 6) & 15;
  if (atype == 0) {
#pragma unroll
    for (int i = 0; i < 2; ++i)
#pragma unroll
      for (int j = 0; j < 2; ++j)
#pragma unroll
        for (int r = 0; r < 8; ++r) {
          int R = m0 + wm * 32 + i * 16 + r + 8 * g;
          int hh = wn * 32 + j * 16 + c;
          int bb = R >> 11, si = R & 2047;
          float v = (acc[i][j][r] * wscale + qbias[head * 64 + hh]) * 0.125f;
          qb[((size_t)(bb * 16 + head) * 2048 + si) * 64 + hh] = (__bf16)v;
        }
  } else if (atype == 1) {
#pragma unroll
    for (int i = 0; i < 2; ++i)
#pragma unroll
      for (int j = 0; j < 2; ++j)
#pragma unroll
        for (int r = 0; r < 8; ++r) {
          int R = m0 + wm * 32 + i * 16 + r + 8 * g;
          int hh = wn * 32 + j * 16 + c;
          int bb = R >> 11, si = R & 2047;
          kb[((size_t)(bb * 16 + head) * 2048 + si) * 64 + hh] =
              (__bf16)(acc[i][j][r] * wscale);
        }
  } else {  // v stored transposed [b][n][h][s] for direct B-frag loads
#pragma unroll
    for (int i = 0; i < 2; ++i)
#pragma unroll
      for (int j = 0; j < 2; ++j)
#pragma unroll
        for (int r = 0; r < 8; ++r) {
          int R = m0 + wm * 32 + i * 16 + r + 8 * g;
          int hh = wn * 32 + j * 16 + c;
          int bb = R >> 11, si = R & 2047;
          vb[((size_t)(bb * 16 + head) * 64 + hh) * 2048 + si] =
              (__bf16)(acc[i][j][r] * wscale);
        }
  }
}

// ---------------------------------------------------------------------------
// Kernel 3: flash attention with Transformer-XL relative positions.
// One wave = 16 query rows; 8 waves per block; no cross-wave sync needed.
// ---------------------------------------------------------------------------
__global__ __launch_bounds__(256) void attn_kernel(const __bf16* __restrict__ qbf,
                                                   const __bf16* __restrict__ kbf,
                                                   const __bf16* __restrict__ vbf,
                                                   const __bf16* __restrict__ ebf,
                                                   __bf16* __restrict__ obf) {
  __shared__ __attribute__((aligned(16))) __bf16 pbuf[8][16 * 32];
  int blk = blockIdx.x;
  int qblk = blk & 15, head = (blk >> 4) & 15, bb = blk >> 8;
  int w = threadIdx.x >> 5;
  int lane = threadIdx.x & 31, c = lane & 15, g = lane >> 4;
  int qw = qblk * 128 + w * 16;

  const __bf16* qb = qbf + (size_t)(bb * 16 + head) * 2048 * 64;
  const __bf16* kb = kbf + (size_t)(bb * 16 + head) * 2048 * 64;
  const __bf16* vb = vbf + (size_t)(bb * 16 + head) * 64 * 2048;
  const __bf16* eb = ebf + (size_t)head * 2048 * 64;
  __bf16* pb = &pbuf[w][0];

  // Q fragments for this wave's 16 rows (h = 0..31, 32..63), reused all loop.
  v16bf aq0 = load_afrag(qb + (size_t)qw * 64, 64);
  v16bf aq1 = load_afrag(qb + (size_t)qw * 64 + 32, 64);

  v8f o0 = zero8(), o1 = zero8(), o2 = zero8(), o3 = zero8();
  float mrow[8], lrow[8];
#pragma unroll
  for (int r = 0; r < 8; ++r) { mrow[r] = -3.0e38f; lrow[r] = 0.0f; }

  for (int k0 = 0; k0 < qw + 16; k0 += 32) {
    if (k0 + 32 < qw + 16) {  // software prefetch of next K/V tiles
      __builtin_prefetch(kb + (size_t)(k0 + 32) * 64, 0, 0);
      __builtin_prefetch(vb + (size_t)c * 2048 + k0 + 32, 0, 0);
    }
    // ---- content scores S = Q * K^T  (two 16x16 tiles over 32 keys) ----
    v8f S0 = zero8(), S1 = zero8();
    S0 = wmma_bf16(aq0, load_bfrag(kb + (size_t)(k0 + 0) * 64 + 0, 64), S0);
    S0 = wmma_bf16(aq1, load_bfrag(kb + (size_t)(k0 + 0) * 64 + 32, 64), S0);
    S1 = wmma_bf16(aq0, load_bfrag(kb + (size_t)(k0 + 16) * 64 + 0, 64), S1);
    S1 = wmma_bf16(aq1, load_bfrag(kb + (size_t)(k0 + 16) * 64 + 32, 64), S1);

    // ---- positional scores T[q][d'] = Q * emb[dbase + d']^T, d' in [0,48) --
    int dbase = qw - k0 - 31;
    v8f T0 = zero8(), T1 = zero8(), T2 = zero8();
    T0 = wmma_bf16(aq0, load_bfrag_eclamp(eb, dbase + 0, 0), T0);
    T0 = wmma_bf16(aq1, load_bfrag_eclamp(eb, dbase + 0, 32), T0);
    T1 = wmma_bf16(aq0, load_bfrag_eclamp(eb, dbase + 16, 0), T1);
    T1 = wmma_bf16(aq1, load_bfrag_eclamp(eb, dbase + 16, 32), T1);
    T2 = wmma_bf16(aq0, load_bfrag_eclamp(eb, dbase + 32, 0), T2);
    T2 = wmma_bf16(aq1, load_bfrag_eclamp(eb, dbase + 32, 32), T2);

    // ---- relative_causal gather (d' = m + 31 - nk) + causal mask ----
#pragma unroll
    for (int r = 0; r < 8; ++r) {
      int m = r + 8 * g;
      int q = qw + m;
      {  // tile 0: keys k0..k0+15, d' in [16,46]
        int dp = m + 31 - c;
        int sl = (dp & 15) | (g << 4);
        float ta = __shfl(T1[r], sl, 32);
        float tb = __shfl(T2[r], sl, 32);
        float s = S0[r] + (dp < 32 ? ta : tb);
        S0[r] = (k0 + c > q) ? -1.0e9f : s;
      }
      {  // tile 1: keys k0+16..k0+31, d' in [0,30]
        int dp = m + 15 - c;
        int sl = (dp & 15) | (g << 4);
        float ta = __shfl(T0[r], sl, 32);
        float tb = __shfl(T1[r], sl, 32);
        float s = S1[r] + (dp < 16 ? ta : tb);
        S1[r] = (k0 + 16 + c > q) ? -1.0e9f : s;
      }
    }

    // ---- online softmax + stage P for the P*V WMMA ----
#pragma unroll
    for (int r = 0; r < 8; ++r) {
      float rm = fmaxf(S0[r], S1[r]);
#pragma unroll
      for (int off = 1; off < 16; off <<= 1) rm = fmaxf(rm, __shfl_xor(rm, off, 32));
      float mnew = fmaxf(mrow[r], rm);
      float fs = __expf(mrow[r] - mnew);
      float p0 = __expf(S0[r] - mnew);
      float p1 = __expf(S1[r] - mnew);
      float rs = p0 + p1;
#pragma unroll
      for (int off = 1; off < 16; off <<= 1) rs += __shfl_xor(rs, off, 32);
      lrow[r] = lrow[r] * fs + rs;
      mrow[r] = mnew;
      o0[r] *= fs; o1[r] *= fs; o2[r] *= fs; o3[r] *= fs;
      int m = r + 8 * g;
      pb[m * 32 + c] = (__bf16)p0;
      pb[m * 32 + 16 + c] = (__bf16)p1;
    }

    // ---- O += P * V  (V stored [h][s]: B-frags straight from global) ----
    v16bf ap = load_afrag(pb, 32);
    o0 = wmma_bf16(ap, load_bfrag(vb + (size_t)0 * 16 * 2048 + k0, 2048), o0);
    o1 = wmma_bf16(ap, load_bfrag(vb + (size_t)1 * 16 * 2048 + k0, 2048), o1);
    o2 = wmma_bf16(ap, load_bfrag(vb + (size_t)2 * 16 * 2048 + k0, 2048), o2);
    o3 = wmma_bf16(ap, load_bfrag(vb + (size_t)3 * 16 * 2048 + k0, 2048), o3);
  }

  // ---- normalize and write o (bf16, [b*s][n*h] row-major for out GEMM) ----
#pragma unroll
  for (int r = 0; r < 8; ++r) {
    float inv = 1.0f / lrow[r];
    int row = qw + r + 8 * g;
    size_t base = ((size_t)(bb * 2048 + row)) * 1024 + head * 64 + c;
    obf[base + 0]  = (__bf16)(o0[r] * inv);
    obf[base + 16] = (__bf16)(o1[r] * inv);
    obf[base + 32] = (__bf16)(o2[r] * inv);
    obf[base + 48] = (__bf16)(o3[r] * inv);
  }
}

// ---------------------------------------------------------------------------
// Kernel 4: output projection  (4096 x 1024) x (1024 x 1024) + bias -> f32
//   K-step 64; A fragments (already bf16) come straight from global.
// ---------------------------------------------------------------------------
__global__ __launch_bounds__(256) void out_gemm(const __bf16* __restrict__ A,
                                                const float* __restrict__ W,
                                                const float* __restrict__ bias,
                                                float* __restrict__ out) {
  __shared__ __attribute__((aligned(16))) __bf16 Bs[64 * 64];
  int m0 = blockIdx.x * 128, n0 = blockIdx.y * 64;
  int t = threadIdx.x;
  int w = t >> 5, wm = w >> 1, wn = w & 1;
  v8f acc[2][2];
#pragma unroll
  for (int i = 0; i < 2; ++i)
#pragma unroll
    for (int j = 0; j < 2; ++j) acc[i][j] = zero8();

  for (int k0 = 0; k0 < 1024; k0 += 64) {
    {  // stage B tile 64x64 transposed to [n][k], packed b32 LDS stores
      int nb = (t & 15) * 4, kb4 = (t >> 4) * 4;
#pragma unroll
      for (int i = 0; i < 4; ++i)
#pragma unroll
        for (int j = 0; j < 2; ++j) {
          float lo = W[(size_t)(k0 + kb4 + 2 * j) * 1024 + n0 + nb + i];
          float hi = W[(size_t)(k0 + kb4 + 2 * j + 1) * 1024 + n0 + nb + i];
          store_pk_bf16(Bs + (nb + i) * 64 + kb4 + 2 * j, lo, hi);
        }
    }
    __syncthreads();
    v16bf a00 = load_afrag(A + (size_t)(m0 + wm * 32 + 0) * 1024 + k0, 1024);
    v16bf a01 = load_afrag(A + (size_t)(m0 + wm * 32 + 0) * 1024 + k0 + 32, 1024);
    v16bf a10 = load_afrag(A + (size_t)(m0 + wm * 32 + 16) * 1024 + k0, 1024);
    v16bf a11 = load_afrag(A + (size_t)(m0 + wm * 32 + 16) * 1024 + k0 + 32, 1024);
    v16bf b00 = load_bfrag(Bs + (wn * 32 + 0) * 64 + 0, 64);
    v16bf b01 = load_bfrag(Bs + (wn * 32 + 0) * 64 + 32, 64);
    v16bf b10 = load_bfrag(Bs + (wn * 32 + 16) * 64 + 0, 64);
    v16bf b11 = load_bfrag(Bs + (wn * 32 + 16) * 64 + 32, 64);
    acc[0][0] = wmma_bf16(a00, b00, acc[0][0]);
    acc[0][0] = wmma_bf16(a01, b01, acc[0][0]);
    acc[0][1] = wmma_bf16(a00, b10, acc[0][1]);
    acc[0][1] = wmma_bf16(a01, b11, acc[0][1]);
    acc[1][0] = wmma_bf16(a10, b00, acc[1][0]);
    acc[1][0] = wmma_bf16(a11, b01, acc[1][0]);
    acc[1][1] = wmma_bf16(a10, b10, acc[1][1]);
    acc[1][1] = wmma_bf16(a11, b11, acc[1][1]);
    __syncthreads();
  }

  int lane = t & 31, c = lane & 15, g = lane >> 4;
  const float oscale = 0.03125f;  // (1024*1024)^-0.25
#pragma unroll
  for (int i = 0; i < 2; ++i)
#pragma unroll
    for (int j = 0; j < 2; ++j)
#pragma unroll
      for (int r = 0; r < 8; ++r) {
        int R = m0 + wm * 32 + i * 16 + r + 8 * g;
        int C = n0 + wn * 32 + j * 16 + c;
        out[(size_t)R * 1024 + C] = acc[i][j][r] * oscale + bias[C];
      }
}

// ---------------------------------------------------------------------------
// Host launcher
// ---------------------------------------------------------------------------
extern "C" void kernel_launch(void* const* d_in, const int* in_sizes, int n_in,
                              void* d_out, int out_size, void* d_ws, size_t ws_size,
                              hipStream_t stream) {
  (void)in_sizes; (void)n_in; (void)out_size; (void)ws_size;
  const float* X   = (const float*)d_in[0];  // (2,2048,1024)
  const float* QKV = (const float*)d_in[1];  // (1024,3,16,64)
  const float* QB  = (const float*)d_in[2];  // (16,64)
  const float* POS = (const float*)d_in[3];  // (256,16,64)
  const float* OW  = (const float*)d_in[4];  // (1024,1024)
  const float* OB  = (const float*)d_in[5];  // (1024,)
  float* OUT = (float*)d_out;                // (2,2048,1024) f32

  char* ws = (char*)d_ws;
  __bf16* qb = (__bf16*)(ws + (size_t)0);          // 8 MB  [b][n][s][h]
  __bf16* kb = (__bf16*)(ws + ((size_t)8 << 20));  // 8 MB  [b][n][s][h]
  __bf16* vb = (__bf16*)(ws + ((size_t)16 << 20)); // 8 MB  [b][n][h][s]
  __bf16* eb = (__bf16*)(ws + ((size_t)24 << 20)); // 4 MB  [n][s][h]
  __bf16* ob = (__bf16*)(ws + ((size_t)28 << 20)); // 8 MB  [b*s][n*h]

  emb_kernel<<<dim3(16 * 2048), dim3(64), 0, stream>>>(POS, eb);
  qkv_gemm<<<dim3(32, 48), dim3(256), 0, stream>>>(X, QKV, QB, qb, kb, vb);
  attn_kernel<<<dim3(512), dim3(256), 0, stream>>>(qb, kb, vb, eb, ob);
  out_gemm<<<dim3(32, 16), dim3(256), 0, stream>>>(ob, OW, OB, OUT);
}
